// MergedSdpaAttention_60644938219556
// MI455X (gfx1250) — matrix operand order repaired and checked by reading, not compiled, not run on hardware
//
#include <hip/hip_runtime.h>
#include <hip/hip_bf16.h>
#include <stdint.h>

// ---------------- problem constants ----------------
#define Q_LEN    2048
#define HIDDEN   4096
#define N_HEADS  32
#define N_KV     8
#define HD       128
#define PAST     1024
#define KV_LEN   3072
#define QKV_N    6144

typedef __attribute__((ext_vector_type(16))) __bf16    bf16x16;
typedef __attribute__((ext_vector_type(8)))  float     floatx8;
typedef __attribute__((ext_vector_type(4)))  uint32_t  u32x4;
typedef __attribute__((ext_vector_type(4)))  int       i32x4;
typedef __attribute__((ext_vector_type(8)))  int       i32x8;

#define GAS __attribute__((address_space(1)))

__device__ __forceinline__ floatx8 wmma_bf16(bf16x16 a, bf16x16 b, floatx8 c) {
  // D = A(16x32 bf16) * B(32x16 bf16) + C(16x16 f32)
  return __builtin_amdgcn_wmma_f32_16x16x32_bf16(false, a, false, b, (short)0, c, false, false);
}

__device__ __forceinline__ uint16_t f2bf(float f) {
  uint32_t u = __float_as_uint(f);
  uint32_t r = u + 0x7FFFu + ((u >> 16) & 1u);
  return (uint16_t)(r >> 16);
}

// global-AS fragment load: two 16B global_load_b128 (LOADcnt only, no flat/DScnt coupling)
__device__ __forceinline__ bf16x16 ld_frag_g(const uint16_t* p0, const uint16_t* p1) {
  union { bf16x16 v; u32x4 u[2]; } f;
  f.u[0] = *(const GAS u32x4*)p0;
  f.u[1] = *(const GAS u32x4*)p1;
  return f.v;
}

// generic-AS fragment load (used for LDS staging -> ds_load_b128)
__device__ __forceinline__ bf16x16 ld_frag(const uint16_t* p0, const uint16_t* p1) {
  union { bf16x16 v; u32x4 u[2]; } f;
  f.u[0] = *(const u32x4*)p0;
  f.u[1] = *(const u32x4*)p1;
  return f.v;
}

__device__ __forceinline__ void st_global_f32(float* p, float v) {
  *(GAS float*)p = v;
}
__device__ __forceinline__ void st_global_u16(uint16_t* p, uint16_t v) {
  *(GAS uint16_t*)p = v;
}

// TDM: DMA a [rows x 128] bf16 tile (row-major, stride 128) from global into LDS.
// D# layout per CDNA5 ISA ch.8: group0 = count/lds_addr/global_addr/type,
// group1 = data_size + tensor dims/strides + tile dims. Groups 2/3 unused (2D tile).
__device__ __forceinline__ void tdm_load_tile_b16(uint32_t lds_off, const uint16_t* gptr,
                                                  uint32_t tensor_rows, uint32_t tile_rows) {
  uint64_t ga = (uint64_t)(uintptr_t)gptr;
  u32x4 g0 = { 1u,                                   // count=1 (valid), gather off
               lds_off,                              // LDS byte address
               (uint32_t)ga,                         // global_addr[31:0]
               (uint32_t)((ga >> 32) & 0x01FFFFFFu) | (2u << 30) };  // addr[56:32] | type=2
  i32x8 g1 = { (int)0x00010000u,                     // wg_mask=0, data_size=1 (2 bytes)
               (int)(HD << 16),                      // tensor_dim0 = 128 (bits 79:48 lo)
               (int)(tensor_rows << 16),             // tensor_dim1 (bits 111:80 lo)
               (int)(HD << 16),                      // tile_dim0 = 128 (bits 127:112)
               (int)tile_rows,                       // tile_dim1 (bits 143:128)
               (int)HD,                              // tensor_dim0_stride = 128 (bits 207:160 lo)
               0, 0 };
  i32x4 gz = { 0, 0, 0, 0 };
  i32x8 gz8 = { 0, 0, 0, 0, 0, 0, 0, 0 };
  __builtin_amdgcn_tensor_load_to_lds(g0, g1, gz, gz, gz8, 0);
}

// ---------------- f32 -> bf16 conversion ----------------
__global__ void cvt_bf16_kernel(const float* __restrict__ in, uint16_t* __restrict__ out, int n) {
  int i = blockIdx.x * blockDim.x + threadIdx.x;
  if (i < n) st_global_u16(out + i, f2bf(*(const GAS float*)(in + i)));
}

// ---------------- GEMM: C[M,N] = A[M,K] * B[N,K]^T (bf16 in, f32 out) ----------------
// one wave per 64x64 output tile; 4x4 grid of 16x16x32 WMMA tiles (~190 VGPRs, no spills)
__global__ void __launch_bounds__(32, 1)
gemm_abT_kernel(const uint16_t* __restrict__ A, const uint16_t* __restrict__ B,
                float* __restrict__ C, int M, int N, int K) {
  const int lane  = threadIdx.x;
  const int lmod  = lane & 15;
  const int lhalf = lane >> 4;
  const int n0 = blockIdx.x * 64;
  const int m0 = blockIdx.y * 64;

  // per-lane row pointers, advanced by 32 elements per K-step
  const uint16_t* ap[4];
  const uint16_t* bp[4];
#pragma unroll
  for (int i = 0; i < 4; ++i) {
    ap[i] = A + (size_t)(m0 + i * 16 + lmod) * K + lhalf * 8;   // A frag: K {0-7,16-23}/{8-15,24-31}
    bp[i] = B + (size_t)(n0 + i * 16 + lmod) * K + lhalf * 8;   // B frag: K {0-7,16-23}/{8-15,24-31}
  }

  const floatx8 z8 = {0.f,0.f,0.f,0.f,0.f,0.f,0.f,0.f};
  floatx8 acc[4][4];
#pragma unroll
  for (int i = 0; i < 4; ++i)
#pragma unroll
    for (int j = 0; j < 4; ++j) acc[i][j] = z8;

  for (int k = 0; k < K; k += 32) {
    bf16x16 af[4], bfr[4];
#pragma unroll
    for (int i = 0; i < 4; ++i) {
      af[i] = ld_frag_g(ap[i], ap[i] + 16);
      __builtin_prefetch((const GAS uint16_t*)(ap[i] + 64), 0, 1);  // global_prefetch_b8
      ap[i] += 32;
    }
#pragma unroll
    for (int j = 0; j < 4; ++j) {
      bfr[j] = ld_frag_g(bp[j], bp[j] + 16);
      __builtin_prefetch((const GAS uint16_t*)(bp[j] + 64), 0, 1);
      bp[j] += 32;
    }
#pragma unroll
    for (int mi = 0; mi < 4; ++mi)
#pragma unroll
      for (int ni = 0; ni < 4; ++ni)
        acc[mi][ni] = wmma_bf16(af[mi], bfr[ni], acc[mi][ni]);
  }

#pragma unroll
  for (int mi = 0; mi < 4; ++mi)
#pragma unroll
    for (int ni = 0; ni < 4; ++ni)
#pragma unroll
      for (int v = 0; v < 8; ++v) {
        int r = m0 + mi * 16 + v + 8 * lhalf;
        int c = n0 + ni * 16 + lmod;
        st_global_f32(C + (size_t)r * N + c, acc[mi][ni][v]);
      }
}

// ---------------- RoPE + pack Q/K/V(as V^T) to bf16 ----------------
__global__ void rope_pack_kernel(const float* __restrict__ qkv,
                                 const float* __restrict__ cosT, const float* __restrict__ sinT,
                                 uint16_t* __restrict__ Qb, uint16_t* __restrict__ Kb,
                                 uint16_t* __restrict__ VTb) {
  int idx = blockIdx.x * blockDim.x + threadIdx.x;
  if (idx >= Q_LEN * QKV_N) return;
  int q = idx / QKV_N;
  int c = idx - q * QKV_N;
  int d = c & (HD - 1);
  float x = *(const GAS float*)(qkv + idx);
  if (c < 5120) {  // rope applies to q and k
    float cs = *(const GAS float*)(cosT + q * HD + d);
    float sn = *(const GAS float*)(sinT + q * HD + d);
    float other = (d < 64) ? -*(const GAS float*)(qkv + idx + 64)
                           :  *(const GAS float*)(qkv + idx - 64);
    x = x * cs + other * sn;
  }
  uint16_t b = f2bf(x);
  if (c < HIDDEN) {                       // Q -> [head][q][d]
    int h = c >> 7;
    st_global_u16(Qb + (size_t)h * Q_LEN * HD + (size_t)q * HD + d, b);
  } else if (c < 5120) {                  // K -> [kv_head][1024+q][d]
    int h = (c - HIDDEN) >> 7;
    st_global_u16(Kb + (size_t)h * KV_LEN * HD + (size_t)(PAST + q) * HD + d, b);
  } else {                                // V -> V^T [kv_head][d][1024+q]
    int h = (c - 5120) >> 7;
    st_global_u16(VTb + (size_t)h * HD * KV_LEN + (size_t)d * KV_LEN + (PAST + q), b);
  }
}

__global__ void pack_past_kernel(const float* __restrict__ pk, const float* __restrict__ pv,
                                 uint16_t* __restrict__ Kb, uint16_t* __restrict__ VTb) {
  int idx = blockIdx.x * blockDim.x + threadIdx.x;
  if (idx >= PAST * N_KV * HD) return;
  int d = idx & (HD - 1);
  int h = (idx >> 7) & (N_KV - 1);
  int p = idx >> 10;  // past_k layout [1024, 8, 128] -> idx = p*1024 + h*128 + d
  st_global_u16(Kb + (size_t)h * KV_LEN * HD + (size_t)p * HD + d,
                f2bf(*(const GAS float*)(pk + idx)));
  st_global_u16(VTb + (size_t)h * HD * KV_LEN + (size_t)d * KV_LEN + p,
                f2bf(*(const GAS float*)(pv + idx)));
}

// ---------------- flash attention: one wave per (head, 16-row q tile) ----------------
// K tiles staged to LDS by the Tensor Data Mover; V fed directly from global.
__global__ void __launch_bounds__(32, 1)
flash_attn_kernel(const uint16_t* __restrict__ Q, const uint16_t* __restrict__ Kall,
                  const uint16_t* __restrict__ VT, uint16_t* __restrict__ O) {
  const int h   = blockIdx.y;
  const int q0  = blockIdx.x * 16;
  const int kvh = h >> 2;  // N_REP = 4
  const int lane  = threadIdx.x;
  const int lmod  = lane & 15;
  const int lhalf = lane >> 4;
  const float scale = 0.08838834764831845f;  // 1/sqrt(128)

  const uint16_t* Qh = Q    + (size_t)h   * Q_LEN  * HD;
  const uint16_t* Kh = Kall + (size_t)kvh * KV_LEN * HD;
  const uint16_t* Vh = VT   + (size_t)kvh * HD * KV_LEN;

  // resident Q fragments: 4 K-steps over head_dim 128
  bf16x16 qf[4];
#pragma unroll
  for (int ks = 0; ks < 4; ++ks) {
    const uint16_t* qp = Qh + (size_t)(q0 + lmod) * HD;
    int ka = ks * 32 + lhalf * 8;
    qf[ks] = ld_frag_g(qp + ka, qp + ka + 16);
  }

  const floatx8 z8 = {0.f,0.f,0.f,0.f,0.f,0.f,0.f,0.f};
  floatx8 o[8];
#pragma unroll
  for (int dt = 0; dt < 8; ++dt) o[dt] = z8;

  float mrow[8], lrow[8];
#pragma unroll
  for (int v = 0; v < 8; ++v) { mrow[v] = -1e30f; lrow[v] = 0.f; }

  __shared__ __align__(16) uint16_t kTile[32 * HD];  // 8 KB TDM staging: [32 kv rows][128 d]
  __shared__ __align__(16) uint16_t pS[16][48];      // P tile staging (C-layout -> A-layout)
  const uint32_t kTileOff = (uint32_t)(uintptr_t)(void*)&kTile[0];  // LDS offset = addr[31:0]

  const int ntiles = (q0 + 16 + PAST + 31) >> 5;  // causal: need kv <= q + PAST

  for (int t = 0; t < ntiles; ++t) {
    const int kv0 = t * 32;

    // ---- async-tensor DMA: K rows [kv0, kv0+32) x 128 into LDS ----
    tdm_load_tile_b16(kTileOff, Kh + (size_t)kv0 * HD, KV_LEN, 32);

    // prefetch next kv tile of V^T row segments while the TDM runs
    if (t + 1 < ntiles)
      __builtin_prefetch((const GAS uint16_t*)(Vh + (size_t)(lane * 4) * KV_LEN + kv0 + 32), 0, 1);

    __builtin_amdgcn_s_wait_tensorcnt(0);

    // ---- S = Q * K^T (1x2 tiles of 16x16, K frags from LDS via ds_load_b128) ----
    floatx8 s[2];
    s[0] = z8; s[1] = z8;
#pragma unroll
    for (int ks = 0; ks < 4; ++ks) {
      bf16x16 kf[2];
#pragma unroll
      for (int ni = 0; ni < 2; ++ni) {
        const uint16_t* kp = &kTile[(ni * 16 + lmod) * HD];
        int kb = ks * 32 + lhalf * 8;
        kf[ni] = ld_frag(kp + kb, kp + kb + 16);
      }
      s[0] = wmma_bf16(qf[ks], kf[0], s[0]);
      s[1] = wmma_bf16(qf[ks], kf[1], s[1]);
    }

    // ---- scale + causal mask + online softmax (row groups = 16 lanes) ----
#pragma unroll
    for (int v = 0; v < 8; ++v) {
      int qg = q0 + v + 8 * lhalf;
      float x0 = s[0][v] * scale;
      float x1 = s[1][v] * scale;
      if (kv0 + lmod      > qg + PAST) x0 = -1e30f;
      if (kv0 + 16 + lmod > qg + PAST) x1 = -1e30f;

      float rm = fmaxf(x0, x1);
#pragma unroll
      for (int d = 1; d < 16; d <<= 1) rm = fmaxf(rm, __shfl_xor(rm, d, 32));
      float mold = mrow[v];
      float mnew = fmaxf(mold, rm);
      float alpha = __expf(mold - mnew);
      mrow[v] = mnew;

      float p0 = __expf(x0 - mnew);
      float p1 = __expf(x1 - mnew);
      float ps = p0 + p1;
#pragma unroll
      for (int d = 1; d < 16; d <<= 1) ps += __shfl_xor(ps, d, 32);
      lrow[v] = lrow[v] * alpha + ps;

#pragma unroll
      for (int dt = 0; dt < 8; ++dt) o[dt][v] *= alpha;

      int r = v + 8 * lhalf;
      pS[r][lmod]      = f2bf(p0);
      pS[r][16 + lmod] = f2bf(p1);
    }
    __syncthreads();

    // ---- O += P * V  (P re-read in A-layout from LDS, V^T rows are K-contiguous) ----
    bf16x16 pf;
    {
      int ka = lhalf * 8;
      pf = ld_frag(&pS[lmod][ka], &pS[lmod][ka + 16]);  // ds_load_b128 x2
    }
#pragma unroll
    for (int dt = 0; dt < 8; ++dt) {
      const uint16_t* vp = Vh + (size_t)(dt * 16 + lmod) * KV_LEN + kv0;
      int kb = lhalf * 8;
      bf16x16 vf = ld_frag_g(vp + kb, vp + kb + 16);
      o[dt] = wmma_bf16(pf, vf, o[dt]);
    }
    __syncthreads();
  }

  // ---- normalize and store (bf16, row-major [q, h*128+d]) ----
#pragma unroll
  for (int v = 0; v < 8; ++v) {
    float inv = 1.0f / lrow[v];
    int qg = q0 + v + 8 * lhalf;
#pragma unroll
    for (int dt = 0; dt < 8; ++dt) {
      int d = dt * 16 + lmod;
      st_global_u16(O + (size_t)qg * HIDDEN + h * HD + d, f2bf(o[dt][v] * inv));
    }
  }
}

// ---------------- host-side orchestration ----------------
extern "C" void kernel_launch(void* const* d_in, const int* in_sizes, int n_in,
                              void* d_out, int out_size, void* d_ws, size_t ws_size,
                              hipStream_t stream) {
  const float* hidden = (const float*)d_in[0];
  const float* cosT   = (const float*)d_in[1];
  const float* sinT   = (const float*)d_in[2];
  const float* past_k = (const float*)d_in[3];
  const float* past_v = (const float*)d_in[4];
  // d_in[5] = attn_mask (causality computed analytically)
  const float* w_qkv  = (const float*)d_in[6];
  const float* w_o    = (const float*)d_in[7];

  uint8_t* w = (uint8_t*)d_ws;
  size_t off = 0;
  uint16_t* Hbf   = (uint16_t*)(w + off); off += (size_t)Q_LEN * HIDDEN * 2;        // also reused as attnOut
  uint16_t* Wqkvb = (uint16_t*)(w + off); off += (size_t)QKV_N * HIDDEN * 2;
  uint16_t* Wob   = (uint16_t*)(w + off); off += (size_t)HIDDEN * HIDDEN * 2;
  float*    qkvF  = (float*)   (w + off); off += (size_t)Q_LEN * QKV_N * 4;
  uint16_t* Qb    = (uint16_t*)(w + off); off += (size_t)N_HEADS * Q_LEN * HD * 2;
  uint16_t* Kb    = (uint16_t*)(w + off); off += (size_t)N_KV * KV_LEN * HD * 2;
  uint16_t* VTb   = (uint16_t*)(w + off); off += (size_t)N_KV * HD * KV_LEN * 2;
  (void)ws_size;

  // 1) convert inputs to bf16
  {
    int n1 = Q_LEN * HIDDEN;
    cvt_bf16_kernel<<<(n1 + 255) / 256, 256, 0, stream>>>(hidden, Hbf, n1);
    int n2 = QKV_N * HIDDEN;
    cvt_bf16_kernel<<<(n2 + 255) / 256, 256, 0, stream>>>(w_qkv, Wqkvb, n2);
    int n3 = HIDDEN * HIDDEN;
    cvt_bf16_kernel<<<(n3 + 255) / 256, 256, 0, stream>>>(w_o, Wob, n3);
  }

  // 2) qkv = hidden @ w_qkv^T   [2048 x 6144]
  {
    dim3 grid(QKV_N / 64, Q_LEN / 64);
    gemm_abT_kernel<<<grid, 32, 0, stream>>>(Hbf, Wqkvb, qkvF, Q_LEN, QKV_N, HIDDEN);
  }

  // 3) RoPE + pack Q/K/V(transposed), plus past KV
  {
    int n = Q_LEN * QKV_N;
    rope_pack_kernel<<<(n + 255) / 256, 256, 0, stream>>>(qkvF, cosT, sinT, Qb, Kb, VTb);
    int np = PAST * N_KV * HD;
    pack_past_kernel<<<(np + 255) / 256, 256, 0, stream>>>(past_k, past_v, Kb, VTb);
  }

  // 4) flash attention -> attnOut (reuses Hbf region; hidden-bf16 is dead now)
  {
    dim3 grid(Q_LEN / 16, N_HEADS);
    flash_attn_kernel<<<grid, 32, 0, stream>>>(Qb, Kb, VTb, Hbf);
  }

  // 5) out = attnOut @ w_o^T  -> f32 d_out [2048 x 4096]
  {
    dim3 grid(HIDDEN / 64, Q_LEN / 64);
    gemm_abT_kernel<<<grid, 32, 0, stream>>>(Hbf, Wob, (float*)d_out, Q_LEN, HIDDEN, HIDDEN);
  }
}
